// Network_30202210025526
// MI455X (gfx1250) — matrix operand order repaired and checked by reading, not compiled, not run on hardware
//
#include <hip/hip_runtime.h>
#include <hip/hip_bf16.h>
#include <cstdint>

typedef __attribute__((ext_vector_type(16))) __bf16 v16bf;
typedef __attribute__((ext_vector_type(8)))  float  v8f;

#define NPL   25600
#define NCR   512
#define CH    256
#define BATCH 256
#define NTAU  8
#define ACTD  40
#define EPPn  409600
#define EPCn  51200
#define ECPn  51200

__device__ __forceinline__ float selu_f(float x) {
  const float lam = 1.0507009873554805f, alp = 1.6732632423543772f;
  return x > 0.f ? lam * x : lam * alp * (expf(x) - 1.f);
}
__device__ __forceinline__ float gelu_f(float x) {
  return 0.5f * x * (1.f + erff(x * 0.7071067811865476f));
}

// ---------------- bf16-WMMA GEMM: C[M,N] = act(A[M,K] @ B + bias) -----------
// B pre-converted to bf16 and transposed: BT[N,K]. One wave -> 16x64 C strip
// (4 accumulators, A fragment reused 4x, 8 WMMAs per unrolled loop body).
__global__ void gemm_wmma(const float* __restrict__ A, const __bf16* __restrict__ BT,
                          const float* __restrict__ bias, float* __restrict__ C,
                          int M, int N, int K, int act)
{
  const int lane  = threadIdx.x;                               // 0..31
  const int tileM = blockIdx.x * 16;
  const int strip = blockIdx.y * blockDim.y + threadIdx.y;     // 64-col strip
  const int n0    = strip * 64;
  const int mrow  = tileM + (lane & 15);
  const int lcol  = (lane & 15);
  const int kbA = (lane < 16) ? 0 : 16;   // A 16x32 frag: lanes0-15 K 0..15
  const int kbB = (lane < 16) ? 0 : 16;   // B 32x16 frag: lanes0-15 K 0..15
  const float* Arow = A + (size_t)mrow * K;

  const __bf16* Bp[4];
  bool nv[4];
#pragma unroll
  for (int t = 0; t < 4; ++t) {
    int nc = n0 + t * 16 + lcol;
    nv[t] = (nc < N);
    Bp[t] = BT + (size_t)(nv[t] ? nc : 0) * K + kbB;
  }

  v8f acc[4];
#pragma unroll
  for (int t = 0; t < 4; ++t) acc[t] = (v8f){0.f,0.f,0.f,0.f,0.f,0.f,0.f,0.f};

#pragma unroll 2
  for (int k0 = 0; k0 < K; k0 += 32) {
    __builtin_prefetch(Arow + k0 + 64, 0, 1);    // global_prefetch_b8
    v16bf a;
    const float* pa0 = Arow + k0 + kbA;
    const float* pa1 = pa0 + 8;
#pragma unroll
    for (int i = 0; i < 8; ++i) {
      a[i]     = (__bf16)pa0[i];
      a[8 + i] = (__bf16)pa1[i];
    }
    v16bf b0 = *(const v16bf*)(Bp[0] + k0);
    v16bf b1 = *(const v16bf*)(Bp[1] + k0);
    v16bf b2 = *(const v16bf*)(Bp[2] + k0);
    v16bf b3 = *(const v16bf*)(Bp[3] + k0);
    acc[0] = __builtin_amdgcn_wmma_f32_16x16x32_bf16(false, a, false, b0, (short)0, acc[0], false, false);
    acc[1] = __builtin_amdgcn_wmma_f32_16x16x32_bf16(false, a, false, b1, (short)0, acc[1], false, false);
    acc[2] = __builtin_amdgcn_wmma_f32_16x16x32_bf16(false, a, false, b2, (short)0, acc[2], false, false);
    acc[3] = __builtin_amdgcn_wmma_f32_16x16x32_bf16(false, a, false, b3, (short)0, acc[3], false, false);
  }

  const int mbase = tileM + ((lane < 16) ? 0 : 8);
#pragma unroll
  for (int t = 0; t < 4; ++t) {
    if (!nv[t]) continue;
    const int nc = n0 + t * 16 + lcol;
    const float bs = bias ? bias[nc] : 0.f;
#pragma unroll
    for (int r = 0; r < 8; ++r) {
      float x = acc[t][r] + bs;
      if (act == 1)      x = fmaxf(x, 0.f);
      else if (act == 2) x = selu_f(x);
      C[(size_t)(mbase + r) * N + nc] = x;
    }
  }
}

// ---------------- small utility kernels ----------------
__global__ void zero_f(float* p, int n) {
  int t = blockIdx.x * blockDim.x + threadIdx.x; if (t < n) p[t] = 0.f;
}
__global__ void fill_u(unsigned* p, unsigned v, int n) {
  int t = blockIdx.x * blockDim.x + threadIdx.x; if (t < n) p[t] = v;
}
// WT[n*K+k] = (bf16) W[k*N+n]
__global__ void transpose_w_bf(const float* __restrict__ W, int K, int N,
                               __bf16* __restrict__ WT) {
  int t = blockIdx.x * blockDim.x + threadIdx.x;
  if (t >= N * K) return;
  int n = t / K, k = t - n * K;
  WT[t] = (__bf16)W[(size_t)k * N + n];
}
// noisy-linear effective weight, transposed bf16: WT[n*K+k] = W[k*N+n] + sW*eW
__global__ void noisy_wt_bf(const float* __restrict__ W, const float* __restrict__ sW,
                            const float* __restrict__ eW, int K, int N,
                            __bf16* __restrict__ WT) {
  int t = blockIdx.x * blockDim.x + threadIdx.x;
  if (t >= N * K) return;
  int n = t / K, k = t - n * K;
  size_t i = (size_t)k * N + n;
  WT[t] = (__bf16)(W[i] + sW[i] * eW[i]);
}
__global__ void noisy_bias(const float* b, const float* sb, const float* eb,
                           int N, float* out) {
  int t = blockIdx.x * blockDim.x + threadIdx.x;
  if (t < N) out[t] = b[t] + sb[t] * eb[t];
}
// fold relation matrix (and optional per-head prior*scale) into projection weight.
// WT[n*K+k] = (bf16) sum_d W[k*256 + h*64 + d] * rel[h,d,e] * sc,  n = h*64+e
__global__ void fuse_rel_w_bf(const float* __restrict__ W, const float* __restrict__ rel,
                              const float* __restrict__ pvec, float scale, int K,
                              __bf16* __restrict__ WT) {
  int t = blockIdx.x * blockDim.x + threadIdx.x;
  if (t >= 256 * K) return;
  int n = t / K, k = t - n * K;
  int h = n >> 6, e = n & 63;
  float sc = scale * (pvec ? pvec[h] : 1.f);
  const float* wrow = W + (size_t)k * 256 + h * 64;
  const float* rcol = rel + (size_t)h * 4096 + e;
  float acc = 0.f;
#pragma unroll 8
  for (int d = 0; d < 64; ++d) acc += wrow[d] * rcol[(size_t)d * 64];
  WT[t] = (__bf16)(acc * sc);
}
__global__ void fuse_rel_b(const float* __restrict__ b, const float* __restrict__ rel,
                           const float* __restrict__ pvec, float scale,
                           float* __restrict__ bout) {
  int n = blockIdx.x * blockDim.x + threadIdx.x;
  if (n >= 256) return;
  int h = n >> 6, e = n & 63;
  float sc = scale * (pvec ? pvec[h] : 1.f);
  const float* brow = b + h * 64;
  const float* rcol = rel + (size_t)h * 4096 + e;
  float acc = 0.f;
#pragma unroll 8
  for (int d = 0; d < 64; ++d) acc += brow[d] * rcol[(size_t)d * 64];
  bout[n] = acc * sc;
}

// ---------------- edge softmax / scatter ----------------
__global__ void edge_logits(const int* __restrict__ ei, int E,
                            const float* __restrict__ q, const float* __restrict__ kr,
                            float* __restrict__ logit, unsigned* __restrict__ mkey) {
  int t = blockIdx.x * blockDim.x + threadIdx.x;
  if (t >= E * 4) return;
  int e = t >> 2, h = t & 3;
  int s = ei[e], d = ei[E + e];
  const float4* qp = (const float4*)(q  + (size_t)d * CH + h * 64);
  const float4* kp = (const float4*)(kr + (size_t)s * CH + h * 64);
  float acc = 0.f;
#pragma unroll
  for (int i = 0; i < 16; ++i) {
    float4 a = qp[i], b = kp[i];
    acc += a.x * b.x + a.y * b.y + a.z * b.z + a.w * b.w;
  }
  logit[t] = acc;
  unsigned bits = __float_as_uint(acc);
  unsigned key = (bits & 0x80000000u) ? ~bits : (bits | 0x80000000u);
  atomicMax(mkey + ((size_t)d * 4 + h), key);
}
__global__ void edge_expsum(const int* __restrict__ ei, int E, float* __restrict__ logit,
                            const unsigned* __restrict__ mkey, float* __restrict__ ssum) {
  int t = blockIdx.x * blockDim.x + threadIdx.x;
  if (t >= E * 4) return;
  int e = t >> 2, h = t & 3;
  int d = ei[E + e];
  unsigned key  = mkey[(size_t)d * 4 + h];
  unsigned bits = (key & 0x80000000u) ? (key ^ 0x80000000u) : ~key;
  float m  = __uint_as_float(bits);
  float wv = expf(logit[t] - m);
  logit[t] = wv;
  atomicAdd(ssum + ((size_t)d * 4 + h), wv);
}
__global__ void edge_scatter(const int* __restrict__ ei, int E,
                             const float* __restrict__ logit, const float* __restrict__ ssum,
                             const float* __restrict__ vr, float* __restrict__ agg) {
  int t = blockIdx.x * blockDim.x + threadIdx.x;
  if (t >= E * 64) return;
  int e = t >> 6, j = t & 63, h = j >> 4, d4 = (j & 15) * 4;
  int s = ei[e], d = ei[E + e];
  float coef = logit[(size_t)e * 4 + h] / (ssum[(size_t)d * 4 + h] + 1e-16f);
  const float4 vv = *(const float4*)(vr + (size_t)s * CH + h * 64 + d4);
  float* ap = agg + (size_t)d * CH + h * 64 + d4;
  atomicAdd(ap + 0, vv.x * coef);
  atomicAdd(ap + 1, vv.y * coef);
  atomicAdd(ap + 2, vv.z * coef);
  atomicAdd(ap + 3, vv.w * coef);
}

// ---------------- elementwise stages ----------------
__global__ void gelu_ip(float* p, int n) {
  int t = blockIdx.x * blockDim.x + threadIdx.x;
  if (t < n) p[t] = gelu_f(p[t]);
}
__global__ void skip_selu_k(const float* __restrict__ o, const float* __restrict__ x,
                            const float* __restrict__ skipP, float* __restrict__ dst, int n) {
  int t = blockIdx.x * blockDim.x + threadIdx.x;
  if (t >= n) return;
  float v = o[t];
  if (x) {
    float a = 1.f / (1.f + expf(-skipP[0]));
    v = a * v + (1.f - a) * x[t];
  }
  dst[t] = selu_f(v);
}
__global__ void pool_k(const float* __restrict__ xp, const float* __restrict__ xc,
                       float* __restrict__ xpool) {
  int t = blockIdx.x * blockDim.x + threadIdx.x;
  if (t >= BATCH * CH) return;
  int b = t >> 8, c = t & 255;
  float acc = 0.f;
  const float* base = xp + ((size_t)b * 100) * CH + c;
#pragma unroll 4
  for (int i = 0; i < 100; ++i) acc += base[(size_t)i * CH];
  acc += xc[((size_t)b * 2) * CH + c] + xc[((size_t)b * 2 + 1) * CH + c];
  xpool[t] = acc;
}
__global__ void cos_k(const float* __restrict__ taus, float* __restrict__ cb) {
  int t = blockIdx.x * blockDim.x + threadIdx.x;
  if (t >= BATCH * NTAU * 64) return;
  int bt = t >> 6, j = t & 63;
  cb[t] = cosf(taus[bt] * 3.14159265358979323846f * (float)(j + 1));
}
__global__ void hmul_k(const float* __restrict__ xpool, const float* __restrict__ cosx,
                       float* __restrict__ hb) {
  int t = blockIdx.x * blockDim.x + threadIdx.x;
  if (t >= BATCH * NTAU * CH) return;
  int bt = t >> 8, b = bt >> 3;
  hb[t] = xpool[(size_t)b * CH + (t & 255)] * cosx[t];
}
__global__ void val_head(const float* __restrict__ h, const float* __restrict__ W,
                         const float* __restrict__ b, const float* __restrict__ sW,
                         const float* __restrict__ sb, const float* __restrict__ eW,
                         const float* __restrict__ eb, float* __restrict__ val) {
  int r = blockIdx.x * blockDim.x + threadIdx.x;
  if (r >= BATCH * NTAU) return;
  const float* hr = h + (size_t)r * CH;
  float acc = 0.f;
#pragma unroll 8
  for (int k = 0; k < CH; ++k) acc += hr[k] * (W[k] + sW[k] * eW[k]);
  val[r] = acc + (b[0] + sb[0] * eb[0]);
}
__global__ void duel_out(const float* __restrict__ adv, const float* __restrict__ val,
                         const float* __restrict__ taus, float* __restrict__ out) {
  int bt = blockIdx.x * blockDim.x + threadIdx.x;
  if (bt >= BATCH * NTAU) return;
  const float* a = adv + (size_t)bt * ACTD;
  float m = 0.f;
#pragma unroll
  for (int i = 0; i < ACTD; ++i) m += a[i];
  m *= (1.f / (float)ACTD);
  float v = val[bt];
  float* o = out + (size_t)bt * ACTD;
#pragma unroll
  for (int i = 0; i < ACTD; ++i) o[i] = v + a[i] - m;
  out[(size_t)BATCH * NTAU * ACTD + bt] = taus[bt];   // append taus
}

// =====================================================================
extern "C" void kernel_launch(void* const* d_in, const int* in_sizes, int n_in,
                              void* d_out, int out_size, void* d_ws, size_t ws_size,
                              hipStream_t stream)
{
  (void)in_sizes; (void)n_in; (void)out_size; (void)ws_size;
  auto F = [&](int i) { return (const float*)d_in[i]; };
  const float* x_plate = F(0);
  const float* x_crane = F(1);
  const int* ei_pp = (const int*)d_in[2];
  const int* ei_pc = (const int*)d_in[3];
  const int* ei_cp = (const int*)d_in[4];
  const float* taus = F(5);

  struct ConvP {
    const float *aWc,*abc,*aWp,*abp,*kWc,*kbc,*kWp,*kbp,*qWc,*qbc,*qWp,*qbp;
    const float *r_cp_a,*r_cp_m,*r_cp_p,*r_pc_a,*r_pc_m,*r_pc_p,*r_pp_a,*r_pp_m,*r_pp_p;
    const float *skc,*skp,*vWc,*vbc,*vWp,*vbp;
  };
  auto loadConv = [&](int b) {
    ConvP p;
    p.aWc=F(b+0);  p.abc=F(b+1);  p.aWp=F(b+2);  p.abp=F(b+3);
    p.kWc=F(b+4);  p.kbc=F(b+5);  p.kWp=F(b+6);  p.kbp=F(b+7);
    p.qWc=F(b+8);  p.qbc=F(b+9);  p.qWp=F(b+10); p.qbp=F(b+11);
    p.r_cp_a=F(b+12); p.r_cp_m=F(b+13); p.r_cp_p=F(b+14);
    p.r_pc_a=F(b+15); p.r_pc_m=F(b+16); p.r_pc_p=F(b+17);
    p.r_pp_a=F(b+18); p.r_pp_m=F(b+19); p.r_pp_p=F(b+20);
    p.skc=F(b+21); p.skp=F(b+22);
    p.vWc=F(b+23); p.vbc=F(b+24); p.vWp=F(b+25); p.vbp=F(b+26);
    return p;
  };
  ConvP c1 = loadConv(10), c2 = loadConv(37);
  const float *advW=F(6), *advb=F(7), *advsW=F(8), *advsb=F(9);
  const float *cosW=F(64), *cosbp=F(65);
  const float *f1W=F(66), *f1b=F(67), *f1sW=F(68), *f1sb=F(69);
  const float *f2W=F(70), *f2b=F(71), *f2sW=F(72), *f2sb=F(73);
  const float *vlW=F(74), *vlb=F(75), *vlsW=F(76), *vlsb=F(77);
  const float *n_advW=F(78), *n_advb=F(79), *n_f1W=F(80), *n_f1b=F(81),
              *n_f2W=F(82), *n_f2b=F(83), *n_vlW=F(84), *n_vlb=F(85);

  // ---- workspace carve (float units; bf16 buffers reuse float-sized slots) ----
  float* W0 = (float*)d_ws;
  size_t off = 0;
  auto alloc = [&](size_t n) { float* p = W0 + off; off += n; return p; };
  float* x_p   = alloc((size_t)NPL*CH); float* x_c   = alloc((size_t)NCR*CH);
  float* q_p   = alloc((size_t)NPL*CH); float* q_c   = alloc((size_t)NCR*CH);
  float* kr_p  = alloc((size_t)NPL*CH); float* vr_p  = alloc((size_t)NPL*CH);
  float* kr_c  = alloc((size_t)NCR*CH); float* vr_c  = alloc((size_t)NCR*CH);
  float* agg_p = alloc((size_t)NPL*CH); float* agg_c = alloc((size_t)NCR*CH);
  float* out_p = alloc((size_t)NPL*CH); float* out_c = alloc((size_t)NCR*CH);
  float* logit = alloc((size_t)EPPn*4);
  float* mkeyf = alloc((size_t)NPL*4);  float* ssum  = alloc((size_t)NPL*4);
  __bf16* WqTp = (__bf16*)alloc(65536); __bf16* WqTc = (__bf16*)alloc(65536);
  __bf16* WaTp = (__bf16*)alloc(65536); __bf16* WaTc = (__bf16*)alloc(65536);
  __bf16* WkrT = (__bf16*)alloc(65536); __bf16* WvrT = (__bf16*)alloc(65536);
  float* bkr  = alloc(256);   float* bvr  = alloc(256);
  __bf16* WcosT= (__bf16*)alloc(16384);
  __bf16* Wt1 = (__bf16*)alloc(65536); float* bt1  = alloc(256);
  __bf16* Wt2 = (__bf16*)alloc(65536); float* bt2  = alloc(256);
  __bf16* WtA = (__bf16*)alloc(10240); float* btA  = alloc(64);
  float* xpool = alloc(65536);
  float* cosbuf= alloc((size_t)2048*64);
  float* cosx  = alloc((size_t)2048*CH);
  float* hb    = alloc((size_t)2048*CH);
  float* h1b   = alloc((size_t)2048*CH);
  float* h2b   = alloc((size_t)2048*CH);
  float* advbf = alloc((size_t)2048*ACTD);
  float* valbf = alloc(2048);
  unsigned* mkey = (unsigned*)mkeyf;

#define EL(kern, n, ...) kern<<<((n)+255)/256, 256, 0, stream>>>(__VA_ARGS__)
  auto gemm = [&](const float* A, const __bf16* BT, const float* bias, float* C,
                  int M, int N, int K, int act) {
    int strips = (N + 63) / 64;
    int nw = strips < 4 ? strips : 4;
    dim3 blk(32, nw);
    dim3 grd(M / 16, (strips + nw - 1) / nw);
    gemm_wmma<<<grd, blk, 0, stream>>>(A, BT, bias, C, M, N, K, act);
  };

  auto run_conv = [&](const float* xpin, const float* xcin, int Kin, const ConvP& P,
                      bool hasSkip) {
    EL(transpose_w_bf, 256*Kin, P.qWp, Kin, 256, WqTp);
    EL(transpose_w_bf, 256*Kin, P.qWc, Kin, 256, WqTc);
    EL(transpose_w_bf, 256*256, P.aWp, 256, 256, WaTp);
    EL(transpose_w_bf, 256*256, P.aWc, 256, 256, WaTc);
    gemm(xpin, WqTp, P.qbp, q_p, NPL, CH, Kin, 0);
    gemm(xcin, WqTc, P.qbc, q_c, NCR, CH, Kin, 0);
    EL(zero_f, NPL*CH, agg_p, NPL*CH);
    EL(zero_f, NCR*CH, agg_c, NCR*CH);

    auto edgeT = [&](const float* xs, int Ns, const float* kW, const float* kb,
                     const float* vW, const float* vb,
                     const float* ra, const float* rm, const float* rp,
                     const int* ei, int E, const float* qd, float* aggd, int Nd,
                     float* krb, float* vrb) {
      EL(fuse_rel_w_bf, 256*Kin, kW, ra, rp, 0.125f, Kin, WkrT);  // 1/sqrt(64)*p folded
      EL(fuse_rel_b, 256,        kb, ra, rp, 0.125f, bkr);
      EL(fuse_rel_w_bf, 256*Kin, vW, rm, (const float*)nullptr, 1.0f, Kin, WvrT);
      EL(fuse_rel_b, 256,        vb, rm, (const float*)nullptr, 1.0f, bvr);
      gemm(xs, WkrT, bkr, krb, Ns, CH, Kin, 0);
      gemm(xs, WvrT, bvr, vrb, Ns, CH, Kin, 0);
      EL(fill_u, Nd*4, mkey, 0x007FFFFFu, Nd*4);  // key(-inf)
      EL(zero_f, Nd*4, ssum, Nd*4);
      EL(edge_logits,  E*4,  ei, E, qd, krb, logit, mkey);
      EL(edge_expsum,  E*4,  ei, E, logit, mkey, ssum);
      EL(edge_scatter, E*64, ei, E, logit, ssum, vrb, aggd);
    };
    // pp: plate->plate, pc: plate->crane, cp: crane->plate (each its own softmax)
    edgeT(xpin, NPL, P.kWp, P.kbp, P.vWp, P.vbp, P.r_pp_a, P.r_pp_m, P.r_pp_p,
          ei_pp, EPPn, q_p, agg_p, NPL, kr_p, vr_p);
    edgeT(xpin, NPL, P.kWp, P.kbp, P.vWp, P.vbp, P.r_pc_a, P.r_pc_m, P.r_pc_p,
          ei_pc, EPCn, q_c, agg_c, NCR, kr_p, vr_p);
    edgeT(xcin, NCR, P.kWc, P.kbc, P.vWc, P.vbc, P.r_cp_a, P.r_cp_m, P.r_cp_p,
          ei_cp, ECPn, q_p, agg_p, NPL, kr_c, vr_c);

    EL(gelu_ip, NPL*CH, agg_p, NPL*CH);
    EL(gelu_ip, NCR*CH, agg_c, NCR*CH);
    gemm(agg_p, WaTp, P.abp, out_p, NPL, CH, CH, 0);
    gemm(agg_c, WaTc, P.abc, out_c, NCR, CH, CH, 0);
    EL(skip_selu_k, NPL*CH, out_p, hasSkip ? xpin : (const float*)nullptr, P.skp, x_p, NPL*CH);
    EL(skip_selu_k, NCR*CH, out_c, hasSkip ? xcin : (const float*)nullptr, P.skc, x_c, NCR*CH);
  };

  run_conv(x_plate, x_crane, 64,  c1, false);  // conv1: 64 -> 256, no skip
  run_conv(x_p,     x_c,     256, c2, true);   // conv2: 256 -> 256, gated skip

  // ---- IQN tail ----
  EL(pool_k, BATCH*CH, x_p, x_c, xpool);
  EL(cos_k, 2048*64, taus, cosbuf);
  EL(transpose_w_bf, 256*64, cosW, 64, 256, WcosT);
  gemm(cosbuf, WcosT, cosbp, cosx, 2048, CH, 64, 1 /*relu*/);
  EL(hmul_k, 2048*CH, xpool, cosx, hb);
  EL(noisy_wt_bf, 256*256, f1W, f1sW, n_f1W, 256, 256, Wt1);
  EL(noisy_bias, 256, f1b, f1sb, n_f1b, 256, bt1);
  gemm(hb, Wt1, bt1, h1b, 2048, CH, CH, 2 /*selu*/);
  EL(noisy_wt_bf, 256*256, f2W, f2sW, n_f2W, 256, 256, Wt2);
  EL(noisy_bias, 256, f2b, f2sb, n_f2b, 256, bt2);
  gemm(h1b, Wt2, bt2, h2b, 2048, CH, CH, 2);
  EL(noisy_wt_bf, ACTD*256, advW, advsW, n_advW, 256, ACTD, WtA);
  EL(noisy_bias, ACTD, advb, advsb, n_advb, ACTD, btA);
  gemm(h2b, WtA, btA, advbf, 2048, ACTD, CH, 0);
  EL(val_head, 2048, h2b, vlW, vlb, vlsW, vlsb, n_vlW, n_vlb, valbf);
  EL(duel_out, 2048, advbf, valbf, taus, (float*)d_out);
#undef EL
}
